// Attention_66855460929912
// MI455X (gfx1250) — matrix operand order repaired and checked by reading, not compiled, not run on hardware
//
#include <hip/hip_runtime.h>
#include <hip/hip_bf16.h>

// ---------------------------------------------------------------------------
// CDNA5 (gfx1250) bf16 WMMA multi-head attention:
//   1) x (f32) -> bf16
//   2) QKV GEMM (WMMA bf16); epilogue scatters Q(scaled by d^-0.5*log2e)/K
//      row-major and V transposed
//   3) flash attention (WMMA bf16, base-2 online softmax, per-lane stats via
//      the S^T/O^T layout trick; single code path, tiny uniform mask branch)
//   4) out-proj GEMM (WMMA bf16) + bias -> f32
// ---------------------------------------------------------------------------

typedef __attribute__((ext_vector_type(16))) __bf16 v16bf;
typedef __attribute__((ext_vector_type(8)))  __bf16 v8bf;
typedef __attribute__((ext_vector_type(4)))  __bf16 v4bf;
typedef __attribute__((ext_vector_type(8)))  float  v8f;

// d^-0.5 * log2(e): scores land directly in the exp2 domain.
#define QSCALE 0.18033688011112042f
// Running-max init and masked-score sentinel are asymmetric on purpose:
// a masked score always sits >= 2e38 below any possible running max, so
// exp2(masked - mn) == 0 in every case (incl. the all-masked first chunk),
// and no explicit zeroing pass is needed.
#define MINIT  (-1.0e38f)
#define MMASK  (-3.0e38f)

__device__ __forceinline__ __bf16 bf(float f) { return (__bf16)f; }

// Build a 16-element bf16 fragment from two contiguous 16-byte chunks.
// Matches the 16-bit A-matrix VGPR layout (two 8-halfword groups).
__device__ __forceinline__ v16bf frag16(const __bf16* p0, const __bf16* p1) {
  v8bf a = *(const v8bf*)p0;
  v8bf b = *(const v8bf*)p1;
  v16bf r;
#pragma unroll
  for (int i = 0; i < 8; i++) { r[i] = a[i]; r[i + 8] = b[i]; }
  return r;
}

__device__ __forceinline__ v8f wmma_bf16(v16bf a, v16bf b, v8f c) {
  // D(16x16 f32) = A(16x32 bf16) * B(32x16 bf16) + C
  return __builtin_amdgcn_wmma_f32_16x16x32_bf16(false, a, false, b,
                                                 (short)0, c, false, false);
}

// ---------------------------------------------------------------------------
// Kernel 1: f32 -> bf16 conversion (vectorized)
// ---------------------------------------------------------------------------
__global__ __launch_bounds__(256) void cvt_f32_to_bf16(
    const float* __restrict__ src, __bf16* __restrict__ dst, int n4) {
  int i = blockIdx.x * 256 + threadIdx.x;
  if (i >= n4) return;
  const float4 v = ((const float4*)src)[i];
  v4bf o;
  o[0] = bf(v.x); o[1] = bf(v.y); o[2] = bf(v.z); o[3] = bf(v.w);
  *(v4bf*)(dst + (size_t)i * 4) = o;
}

// ---------------------------------------------------------------------------
// Kernels 2 & 4: WMMA GEMM, C[4096,N] = A[4096,1024](bf16) * B[1024,N](f32)
// Tile: 128(M) x 128(N) per 256-thread workgroup (8 waves, 4x2 wave grid,
// each wave owns 32x64 = 2x4 16x16 accumulators). K staged in chunks of 32.
// EPI==0: QKV scatter epilogue.  EPI==1: bias + f32 store epilogue.
// ---------------------------------------------------------------------------
template <int EPI>
__global__ __launch_bounds__(256) void gemm128(
    const __bf16* __restrict__ A,     // [4096,1024] row-major bf16
    const float*  __restrict__ Bw,    // [1024,N]    row-major f32
    int N,
    __bf16* __restrict__ Qo, __bf16* __restrict__ Ko, __bf16* __restrict__ Vto,
    float* __restrict__ out, const float* __restrict__ bias) {
  __shared__ __bf16 sA[128 * 40];   // 128 rows x 32 k (stride 40 for bank spread)
  __shared__ __bf16 sB[128 * 40];   // 128 n-rows x 32 k (B staged transposed)

  const int tid  = threadIdx.x;
  const int m0   = blockIdx.y * 128;
  const int n0   = blockIdx.x * 128;
  const int lane = tid & 31, half = lane >> 4, lr = lane & 15;
  const int wave = tid >> 5;
  const int wm   = wave >> 1;   // 0..3 -> 32-row strip
  const int wn   = wave & 1;    // 0..1 -> 64-col strip

  v8f acc[2][4] = {};

  for (int k0 = 0; k0 < 1024; k0 += 32) {
    // Stage A tile: 128x32 bf16, 16B chunks.
#pragma unroll
    for (int i = 0; i < 2; i++) {
      int cid = tid + i * 256;          // 0..511
      int row = cid >> 2;
      int cc  = (cid & 3) << 3;
      *(v8bf*)(sA + row * 40 + cc) =
          *(const v8bf*)(A + (size_t)(m0 + row) * 1024 + k0 + cc);
    }
    // Stage B tile transposed: W[k0..k0+31][n0..n0+127] f32 -> sB[n][k] bf16.
#pragma unroll
    for (int i = 0; i < 4; i++) {
      int cid = tid + i * 256;          // 0..1023
      int kk  = cid >> 5;               // 0..31
      int nn  = (cid & 31) << 2;        // 0..124
      const float4 w = *(const float4*)(Bw + (size_t)(k0 + kk) * N + n0 + nn);
      sB[(nn + 0) * 40 + kk] = bf(w.x);
      sB[(nn + 1) * 40 + kk] = bf(w.y);
      sB[(nn + 2) * 40 + kk] = bf(w.z);
      sB[(nn + 3) * 40 + kk] = bf(w.w);
    }
    __syncthreads();

    v16bf af[2], bfr[4];
#pragma unroll
    for (int mf = 0; mf < 2; mf++) {
      // A-frag: lane = row, k = half*16 + e (contiguous 16)
      const __bf16* pr = sA + (wm * 32 + mf * 16 + lr) * 40;
      af[mf] = frag16(pr + half * 16, pr + half * 16 + 8);
    }
#pragma unroll
    for (int nf = 0; nf < 4; nf++) {
      // B-frag: lane = column, k = half*16 + e (contiguous 16)
      const __bf16* pr = sB + (wn * 64 + nf * 16 + lr) * 40 + half * 16;
      bfr[nf] = frag16(pr, pr + 8);
    }
#pragma unroll
    for (int mf = 0; mf < 2; mf++)
#pragma unroll
      for (int nf = 0; nf < 4; nf++)
        acc[mf][nf] = wmma_bf16(af[mf], bfr[nf], acc[mf][nf]);
    __syncthreads();
  }

  // Epilogue. C layout: lane holds column n = lr, rows m = half*8 + r.
#pragma unroll
  for (int mf = 0; mf < 2; mf++) {
#pragma unroll
    for (int nf = 0; nf < 4; nf++) {
      const int colg = n0 + wn * 64 + nf * 16 + lr;
      const int rowb = m0 + wm * 32 + mf * 16 + half * 8;
      if (EPI == 0) {
        const int which = colg >> 10;        // 0=Q 1=K 2=V
        const int cd    = colg & 1023;
        const int h     = cd >> 6, dcol = cd & 63;
        const int tok   = rowb & 2047;
        const size_t bh = (size_t)((rowb >> 11) * 16 + h);
        if (which == 0) {
          __bf16* p = Qo + (bh * 2048 + tok) * 64 + dcol;
#pragma unroll
          for (int r = 0; r < 8; r++) p[r * 64] = bf(acc[mf][nf][r] * QSCALE);
        } else if (which == 1) {
          __bf16* p = Ko + (bh * 2048 + tok) * 64 + dcol;
#pragma unroll
          for (int r = 0; r < 8; r++) p[r * 64] = bf(acc[mf][nf][r]);
        } else {
          v8bf o;
#pragma unroll
          for (int r = 0; r < 8; r++) o[r] = bf(acc[mf][nf][r]);
          *(v8bf*)(Vto + (bh * 64 + dcol) * 2048 + tok) = o;  // V^T layout
        }
      } else {
        const float bv = bias[colg];
#pragma unroll
        for (int r = 0; r < 8; r++)
          out[(size_t)(rowb + r) * N + colg] = acc[mf][nf][r] + bv;
      }
    }
  }
}

// ---------------------------------------------------------------------------
// Kernel 3: flash attention per (b,h). Block = 4 waves x 16 queries = 64 rows.
// S^T = K * Q^T puts the query in the lane index -> per-lane softmax state.
// O^T = V^T * P^T keeps the query in the lane index for the rescale too.
// Single code path; masking and the O-rescale are small in-place uniform
// branches so no register-set duplication across the loop latch.
// ---------------------------------------------------------------------------
__global__ __launch_bounds__(128) void flash_attn(
    const __bf16* __restrict__ Q, const __bf16* __restrict__ K,
    const __bf16* __restrict__ Vt, const unsigned char* __restrict__ mask,
    __bf16* __restrict__ att) {
  const int bh   = blockIdx.x;                 // 0..31 = b*16+h
  const int wave = threadIdx.x >> 5;
  const int lane = threadIdx.x & 31, half = lane >> 4, lq = lane & 15;
  const int q0   = blockIdx.y * 64 + wave * 16;

  const __bf16* Qh = Q  + (size_t)bh * 2048 * 64;
  const __bf16* Kh = K  + (size_t)bh * 2048 * 64;
  const __bf16* Vh = Vt + (size_t)bh * 64 * 2048;
  const unsigned char* mb = mask + (size_t)(bh >> 4) * 2048;

  // Q^T B-fragments (lane = query column): contiguous 32B per lane.
  v16bf qf[2];
#pragma unroll
  for (int c = 0; c < 2; c++) {
    const __bf16* p = Qh + (size_t)(q0 + lq) * 64 + c * 32 + half * 16;
    qf[c] = frag16(p, p + 8);
  }

  v8f ot[4] = {};               // O^T accumulators: lane = query, rows = feats
  float mi = MINIT, li = 0.f;   // per-lane (per-query) base-2 softmax state

#pragma unroll 1
  for (int key0 = 0; key0 < 2048; key0 += 32) {
    // Key mask for this 32-key chunk as a uniform 32-bit ballot (SGPR).
    const unsigned mbits = __builtin_amdgcn_ballot_w32(mb[key0 + lane] != 0);
    if (key0 + 32 < 2048) {
      __builtin_prefetch(Kh + (size_t)(key0 + 32 + lq) * 64, 0, 0);
      __builtin_prefetch(Vh + (size_t)lq * 2048 + key0 + 32, 0, 0);
    }

    // S^T(16 keys x 16 queries) per 16-key tile j: A = K tile, B = Q^T.
    v8f st[2] = {};
#pragma unroll
    for (int j = 0; j < 2; j++) {
#pragma unroll
      for (int c = 0; c < 2; c++) {
        const __bf16* p = Kh + (size_t)(key0 + j * 16 + lq) * 64 + c * 32;
        v16bf kf = frag16(p + half * 16, p + half * 16 + 8);
        st[j] = wmma_bf16(kf, qf[c], st[j]);
      }
    }

    // Uniform branch, in-place update: no bulk register merges when skipped.
    if (mbits != 0xffffffffu) {
      const unsigned mb0 = (mbits >> (half * 8)) & 0xffu;
      const unsigned mb1 = (mbits >> (16 + half * 8)) & 0xffu;
#pragma unroll
      for (int r = 0; r < 8; r++) {
        if (!((mb0 >> r) & 1u)) st[0][r] = MMASK;
        if (!((mb1 >> r) & 1u)) st[1][r] = MMASK;
      }
    }

    // Lane holds query lq; keys half*8+r (tile 0) and 16+half*8+r (tile 1).
    float cm = MMASK;
#pragma unroll
    for (int r = 0; r < 8; r++) cm = fmaxf(cm, fmaxf(st[0][r], st[1][r]));
    cm = fmaxf(cm, __shfl_xor(cm, 16));          // merge the two key-halves
    const float mn    = fmaxf(mi, cm);
    const float alpha = __builtin_amdgcn_exp2f(mi - mn);   // base-2 domain
    float p0[8], p1[8];
    float ls = 0.f;
#pragma unroll
    for (int r = 0; r < 8; r++) {
      // Masked entries give exp2(<= -2e38) == 0; no explicit zeroing needed.
      p0[r] = __builtin_amdgcn_exp2f(st[0][r] - mn);
      p1[r] = __builtin_amdgcn_exp2f(st[1][r] - mn);
      ls += p0[r] + p1[r];
    }
    ls += __shfl_xor(ls, 16);
    li = li * alpha + ls;

    // O^T rescale only if some lane's running max moved (uniform, in-place).
    if (__builtin_amdgcn_ballot_w32(mn > mi) != 0u) {
#pragma unroll
      for (int t = 0; t < 4; t++)
#pragma unroll
        for (int r = 0; r < 8; r++) ot[t][r] *= alpha;
    }
    mi = mn;

    // Build P^T B-fragment: element e needs key half*16+e -> swap halves.
    v16bf pf;
#pragma unroll
    for (int r = 0; r < 8; r++) {
      const float xa = __shfl_xor(p0[r], 16);
      const float xb = __shfl_xor(p1[r], 16);
      pf[r]     = bf(half ? xb    : p0[r]);
      pf[r + 8] = bf(half ? p1[r] : xa);
    }

    // O^T += V^T(16 feats x 32 keys) * P^T(32 keys x 16 queries)
#pragma unroll
    for (int t = 0; t < 4; t++) {
      const __bf16* p = Vh + (size_t)(t * 16 + lq) * 2048 + key0;
      v16bf vf = frag16(p + half * 16, p + half * 16 + 8);
      ot[t] = wmma_bf16(vf, pf, ot[t]);
    }
  }

  const float inv = (li > 0.f) ? 1.f / li : 0.f;
  const int b = bh >> 4, h = bh & 15;
  const size_t row = (size_t)b * 2048 + q0 + lq;      // lane = query
#pragma unroll
  for (int t = 0; t < 4; t++) {
    v8bf o;
#pragma unroll
    for (int r = 0; r < 8; r++) o[r] = bf(ot[t][r] * inv);
    *(v8bf*)(att + row * 1024 + h * 64 + t * 16 + half * 8) = o;
  }
}

// ---------------------------------------------------------------------------
extern "C" void kernel_launch(void* const* d_in, const int* in_sizes, int n_in,
                              void* d_out, int out_size, void* d_ws, size_t ws_size,
                              hipStream_t stream) {
  (void)in_sizes; (void)n_in; (void)out_size; (void)ws_size;
  const float* x            = (const float*)d_in[0];
  const unsigned char* mask = (const unsigned char*)d_in[1];
  const float* Wqkv         = (const float*)d_in[2];
  const float* Wout         = (const float*)d_in[3];
  const float* bout         = (const float*)d_in[4];
  float* out = (float*)d_out;

  char* ws = (char*)d_ws;
  __bf16* xb  = (__bf16*)(ws);                       // x as bf16   (8 MiB)
  __bf16* Qb  = (__bf16*)(ws + ((size_t)8  << 20));  // Q [bh][n][64], pre-scaled
  __bf16* Kb  = (__bf16*)(ws + ((size_t)16 << 20));  // K [bh][n][64]
  __bf16* Vtb = (__bf16*)(ws + ((size_t)24 << 20));  // V^T [bh][64][n]
  __bf16* att = (__bf16*)(ws + ((size_t)32 << 20));  // attn out [4096][1024]

  // 1) x -> bf16
  cvt_f32_to_bf16<<<4096, 256, 0, stream>>>(x, xb, (4096 * 1024) / 4);
  // 2) QKV projection (scatter into Q/K/V^T, Q scaled by d^-0.5 * log2e)
  gemm128<0><<<dim3(3072 / 128, 4096 / 128), 256, 0, stream>>>(
      xb, Wqkv, 3072, Qb, Kb, Vtb, nullptr, nullptr);
  // 3) flash attention
  flash_attn<<<dim3(32, 2048 / 64), 128, 0, stream>>>(Qb, Kb, Vtb, mask, att);
  // 4) output projection + bias
  gemm128<1><<<dim3(1024 / 128, 4096 / 128), 256, 0, stream>>>(
      att, Wout, 1024, nullptr, nullptr, nullptr, out, bout);
}
